// RNN_20126216750052
// MI455X (gfx1250) — compile-verified, matrix operand order checked by reading, not run on hardware
//
#include <hip/hip_runtime.h>
#include <hip/hip_bf16.h>

#define VOCAB 4096
#define HID   512
#define BB    32
#define TT    512

typedef __attribute__((ext_vector_type(16))) __bf16 v16bf;
typedef __attribute__((ext_vector_type(8)))  float  v8f;

union V16U {
    v16bf v;
    uint4 q[2];
    unsigned short s[16];
};

// float -> bf16, round-to-nearest-even
static __device__ __forceinline__ unsigned short f2bf(float f) {
    unsigned int u = __float_as_uint(f);
    unsigned int r = u + 0x7FFFu + ((u >> 16) & 1u);
    return (unsigned short)(r >> 16);
}

// -----------------------------------------------------------------------------
// Pack a row-major f32 (HID x ncols) weight matrix into bf16 in the WMMA
// B-matrix per-lane register layout:
//   out[((ntile*16 + kstep)*32 + lane)*16 + e] = bf16( W[k][ntile*16 + lane%16] )
//   where half = lane/16,
//         k = kstep*32 + (e<8 ? half*8+e : 16 + half*8 + (e-8))
// Each lane's 16 bf16 fragment is then one contiguous 32-byte load.
// -----------------------------------------------------------------------------
__global__ void pack_wmmaB_bf16(const float* __restrict__ W,
                                unsigned short* __restrict__ out,
                                int ncols) {
    int id = blockIdx.x * 256 + threadIdx.x;
    int total = HID * ncols;
    if (id >= total) return;
    int e     = id & 15;
    int lane  = (id >> 4) & 31;
    int ks    = (id >> 9) & 15;
    int ntile = id >> 13;
    int half  = lane >> 4;
    int k = ks * 32 + ((e < 8) ? (half * 8 + e) : (16 + half * 8 + (e - 8)));
    int n = ntile * 16 + (lane & 15);
    out[id] = f2bf(W[(size_t)k * ncols + n]);
}

// -----------------------------------------------------------------------------
// Sequential recurrence: one workgroup, 32 waves. H lives in LDS as bf16
// row-major (32 x 512). Wave w -> mtile = w/16 (0..1), ntiles (w%16)*2 + {0,1}.
// Per step: 16 K-steps x 2 tiles = 32 v_wmma_f32_16x16x32_bf16 per wave.
// -----------------------------------------------------------------------------
__global__ __launch_bounds__(1024) void rnn_recurrence(
    const int*   __restrict__ X,        // (B, T) tokens
    const float* __restrict__ state,    // (B, HID)
    const float* __restrict__ W_xh,     // (VOCAB, HID)
    const float* __restrict__ b_h,      // (HID)
    const unsigned short* __restrict__ Whh_pk, // packed bf16 B-layout
    unsigned short* __restrict__ H_all, // (T*B, HID) bf16 row-major
    float* __restrict__ H_final)        // (B, HID) f32
{
    __shared__ unsigned short Hs[BB * HID]; // 32 KB

    int tid = threadIdx.x;
    for (int i = tid; i < BB * HID; i += 1024) Hs[i] = f2bf(state[i]);
    __syncthreads();

    const int lane  = tid & 31;
    const int wave  = tid >> 5;
    const int mtile = wave >> 4;          // 0..1
    const int nt0   = (wave & 15) * 2;    // 0..30 (16-col tiles)
    const int mrow  = lane & 15;
    const int half  = lane >> 4;
    const int a_row = mtile * 16 + mrow;  // A-matrix row for this lane

    const int ncol0 = nt0 * 16 + mrow;
    const int ncol1 = ncol0 + 16;
    const float bh0 = b_h[ncol0];
    const float bh1 = b_h[ncol1];

    for (int t = 0; t < TT; ++t) {
        v8f acc0 = {};
        v8f acc1 = {};
#pragma unroll 4
        for (int ks = 0; ks < 16; ++ks) {
            V16U a;
            const unsigned short* hrow = &Hs[a_row * HID + ks * 32 + half * 8];
            a.q[0] = *(const uint4*)(hrow);
            a.q[1] = *(const uint4*)(hrow + 16);
            v16bf b0 = *(const v16bf*)(Whh_pk + (size_t)(((nt0 + 0) * 16 + ks) * 32 + lane) * 16);
            v16bf b1 = *(const v16bf*)(Whh_pk + (size_t)(((nt0 + 1) * 16 + ks) * 32 + lane) * 16);
            acc0 = __builtin_amdgcn_wmma_f32_16x16x32_bf16(false, a.v, false, b0,
                                                           (short)0, acc0, false, false);
            acc1 = __builtin_amdgcn_wmma_f32_16x16x32_bf16(false, a.v, false, b1,
                                                           (short)0, acc1, false, false);
        }
        __syncthreads();  // everyone done reading Hs for step t

        // Fused epilogue: one-hot gather + bias + tanh, write back H.
#pragma unroll
        for (int i = 0; i < 8; ++i) {
            int brow = mtile * 16 + half * 8 + i;       // C layout: M = 8*half + i
            int tok  = X[brow * TT + t];
            float h0 = tanhf(acc0[i] + W_xh[(size_t)tok * HID + ncol0] + bh0);
            float h1 = tanhf(acc1[i] + W_xh[(size_t)tok * HID + ncol1] + bh1);
            unsigned short hb0 = f2bf(h0), hb1 = f2bf(h1);
            Hs[brow * HID + ncol0] = hb0;
            Hs[brow * HID + ncol1] = hb1;
            size_t r = (size_t)(t * BB + brow) * HID;
            H_all[r + ncol0] = hb0;
            H_all[r + ncol1] = hb1;
            if (t == TT - 1) {      // uniform branch
                H_final[brow * HID + ncol0] = h0;
                H_final[brow * HID + ncol1] = h1;
            }
        }
        __syncthreads();  // Hs fully updated for step t+1
    }
}

// -----------------------------------------------------------------------------
// Output GEMM: Y = H_all(bf16, 16384x512) @ W_hq(bf16 packed, 512x4096) + b_q.
// Block = 256 threads (8 waves); wave computes a 16x64 slab (4 tiles),
// grid = (16384/16, 4096/512). A fragment (two b128 loads) reused 4x.
// -----------------------------------------------------------------------------
__global__ __launch_bounds__(256) void rnn_output_gemm(
    const unsigned short* __restrict__ H_all,  // (T*B, HID) bf16 row-major
    const unsigned short* __restrict__ Whq_pk, // packed bf16 B-layout
    const float* __restrict__ b_q,             // (VOCAB)
    float* __restrict__ Y)                     // (T*B, VOCAB) f32
{
    const int lane    = threadIdx.x & 31;
    const int wave    = threadIdx.x >> 5;
    const int mtile   = blockIdx.x;                    // 0..1023
    const int nt_base = blockIdx.y * 32 + wave * 4;    // 16-col tile index, 0..252
    const int mrow    = lane & 15;
    const int half    = lane >> 4;

    const unsigned short* arow = H_all + (size_t)(mtile * 16 + mrow) * HID + half * 8;

    v8f acc[4] = {v8f{}, v8f{}, v8f{}, v8f{}};
#pragma unroll 2
    for (int ks = 0; ks < 16; ++ks) {
        V16U a;
        const unsigned short* ap = arow + ks * 32;
        a.q[0] = *(const uint4*)(ap);
        a.q[1] = *(const uint4*)(ap + 16);
#pragma unroll
        for (int j = 0; j < 4; ++j) {
            v16bf b = *(const v16bf*)(Whq_pk +
                        (size_t)(((nt_base + j) * 16 + ks) * 32 + lane) * 16);
            acc[j] = __builtin_amdgcn_wmma_f32_16x16x32_bf16(false, a.v, false, b,
                                                             (short)0, acc[j], false, false);
        }
    }

#pragma unroll
    for (int j = 0; j < 4; ++j) {
        int ncol = (nt_base + j) * 16 + mrow;   // fixed per lane across all 8 rows
        float bq = b_q[ncol];
#pragma unroll
        for (int i = 0; i < 8; ++i) {
            int r = mtile * 16 + half * 8 + i;
            Y[(size_t)r * VOCAB + ncol] = acc[j][i] + bq;
        }
    }
}

// -----------------------------------------------------------------------------
extern "C" void kernel_launch(void* const* d_in, const int* in_sizes, int n_in,
                              void* d_out, int out_size, void* d_ws, size_t ws_size,
                              hipStream_t stream) {
    const int*   X     = (const int*)  d_in[0];
    const float* state = (const float*)d_in[1];
    const float* W_xh  = (const float*)d_in[2];
    const float* W_hh  = (const float*)d_in[3];
    const float* b_h   = (const float*)d_in[4];
    const float* W_hq  = (const float*)d_in[5];
    const float* b_q   = (const float*)d_in[6];

    float* Y       = (float*)d_out;                       // (T*B, VOCAB)
    float* H_final = Y + (size_t)TT * BB * VOCAB;         // (B, HID)

    char* ws = (char*)d_ws;
    unsigned short* Whh_pk = (unsigned short*)ws;                               // 512 KB
    unsigned short* Whq_pk = (unsigned short*)(ws + 512u * 1024u);              // 4 MB
    unsigned short* H_all  = (unsigned short*)(ws + 512u * 1024u + 4u * 1024u * 1024u); // 16 MB

    // 1) Pack weights into bf16 WMMA B-matrix layout (L2-resident thereafter).
    pack_wmmaB_bf16<<<(HID * HID + 255) / 256, 256, 0, stream>>>(W_hh, Whh_pk, HID);
    pack_wmmaB_bf16<<<(HID * VOCAB + 255) / 256, 256, 0, stream>>>(W_hq, Whq_pk, VOCAB);

    // 2) Sequential recurrence (single workgroup, H in LDS).
    rnn_recurrence<<<1, 1024, 0, stream>>>(X, state, W_xh, b_h, Whh_pk, H_all, H_final);

    // 3) Big parallel output GEMM + bias.
    rnn_output_gemm<<<dim3(TT * BB / 16, VOCAB / 512), 256, 0, stream>>>(
        H_all, Whq_pk, b_q, Y);
}
